// Conv2dAttnBlock_7567732376025
// MI455X (gfx1250) — matrix operand order repaired and checked by reading, not compiled, not run on hardware
//
#include <hip/hip_runtime.h>
#include <hip/hip_bf16.h>
#include <math.h>

typedef __attribute__((ext_vector_type(16))) _Float16 v16h;
typedef __attribute__((ext_vector_type(8)))  _Float16 v8h;
typedef __attribute__((ext_vector_type(8)))  float    v8f;
typedef __attribute__((ext_vector_type(4)))  unsigned int u32x4;
typedef __attribute__((ext_vector_type(8)))  int i32x8;
typedef __attribute__((ext_vector_type(4)))  int i32x4;

static constexpr int BB  = 4;
static constexpr int C   = 512;
static constexpr int N   = 4096;          // 64*64 pixels
static constexpr int G   = 32;
static constexpr int CPG = C / G;         // 16
static constexpr float EPS    = 1e-6f;
static constexpr float QSCALE = 0.04419417382415922f;  // 512^-0.5

#define DEVINL __device__ __forceinline__

// ---- WMMA fragment loaders (CDNA5 16x16x32 f16 layouts) --------------------
// A (16x32, MxK): lane l holds row m=l%16; K chunk base kb=(l/16)*8;
//   VGPR0..3 = K[kb..kb+7], VGPR4..7 = K[kb+16..kb+23]  -> two 16B loads.
DEVINL v16h frag_a(const _Float16* p, int ld) {
  int lane = threadIdx.x & 31;
  const _Float16* q = p + (size_t)(lane & 15) * ld + ((lane >> 4) << 3);
  v8h lo = *(const v8h*)(q);
  v8h hi = *(const v8h*)(q + 16);
  v16h f;
#pragma unroll
  for (int i = 0; i < 8; ++i) { f[i] = lo[i]; f[i + 8] = hi[i]; }
  return f;
}

// B (32x16, KxN) loaded from Bt stored row-major as (n, k): lane l holds
// col n=l%16; K chunk base kb=(l/16)*8: K[kb..kb+7], K[kb+16..kb+23] -> two 16B loads.
DEVINL v16h frag_b(const _Float16* p, int ld) {
  int lane = threadIdx.x & 31;
  const _Float16* q = p + (size_t)(lane & 15) * ld + ((lane >> 4) << 3);
  v8h lo = *(const v8h*)(q);
  v8h hi = *(const v8h*)(q + 16);
  v16h f;
#pragma unroll
  for (int i = 0; i < 8; ++i) { f[i] = lo[i]; f[i + 8] = hi[i]; }
  return f;
}

DEVINL v8f wmma_f16(v16h a, v16h b, v8f c) {
  return __builtin_amdgcn_wmma_f32_16x16x32_f16(false, a, false, b,
                                                (short)0, c, false, false);
}

DEVINL v8f vzero8() { v8f z = {0.f,0.f,0.f,0.f,0.f,0.f,0.f,0.f}; return z; }

// ---- weight f32 -> f16 conversion ------------------------------------------
__global__ void __launch_bounds__(256)
prep_weights_kernel(const float* qw, const float* kw, const float* vw, const float* pw,
                    _Float16* wq, _Float16* wk, _Float16* wv, _Float16* wp) {
  int i = blockIdx.x * 256 + threadIdx.x;
  if (i < C * C) {
    wq[i] = (_Float16)qw[i];
    wk[i] = (_Float16)kw[i];
    wv[i] = (_Float16)vw[i];
    wp[i] = (_Float16)pw[i];
  }
}

// ---- GroupNorm: writes y transposed (N, C) in f16 --------------------------
__global__ void __launch_bounds__(256)
groupnorm_kernel(const float* x, const float* gamma, const float* beta, _Float16* y_t) {
  int b = blockIdx.x / G, g = blockIdx.x % G;
  const float* xp = x + (size_t)b * C * N + (size_t)g * CPG * N;
  __shared__ float red[256];
  int tid = threadIdx.x;
  constexpr int E = CPG * N;  // 65536
  float s = 0.f, ss = 0.f;
  for (int e = tid; e < E; e += 256) { float v = xp[e]; s += v; ss += v * v; }
  red[tid] = s; __syncthreads();
  for (int o = 128; o > 0; o >>= 1) { if (tid < o) red[tid] += red[tid + o]; __syncthreads(); }
  float mean = red[0] / (float)E;
  __syncthreads();
  red[tid] = ss; __syncthreads();
  for (int o = 128; o > 0; o >>= 1) { if (tid < o) red[tid] += red[tid + o]; __syncthreads(); }
  float var  = red[0] / (float)E - mean * mean;
  float rstd = rsqrtf(var + EPS);
  _Float16* yb = y_t + (size_t)b * N * C;
  for (int e = tid; e < E; e += 256) {
    int cl = e >> 12, n = e & (N - 1);
    int c = g * CPG + cl;
    float v = (xp[e] - mean) * rstd * gamma[c] + beta[c];
    yb[(size_t)n * C + c] = (_Float16)v;
  }
}

// ---- QKV projection GEMM ----------------------------------------------------
__global__ void __launch_bounds__(256)
qkv_gemm_kernel(const _Float16* y_t,
                const _Float16* wq, const _Float16* wk, const _Float16* wv,
                const float* qb, const float* kbias, const float* vb,
                _Float16* q_t, _Float16* k_t, _Float16* v_cn) {
  int b = blockIdx.y, mode = blockIdx.z;
  int bm = blockIdx.x / (C / 64), bn = blockIdx.x % (C / 64);
  int wid = threadIdx.x >> 5, lane = threadIdx.x & 31;
  int wm = wid & 3, wn = wid >> 2;
  int r0 = bm * 128 + wm * 32;
  int c0 = bn * 64  + wn * 32;
  const _Float16* A  = y_t + (size_t)b * N * C;
  const _Float16* Wt = (mode == 0) ? wq : (mode == 1 ? wk : wv);
  const float* bias  = (mode == 0) ? qb : (mode == 1 ? kbias : vb);

  v8f acc[2][2] = {{vzero8(), vzero8()}, {vzero8(), vzero8()}};
#pragma unroll 4
  for (int k0 = 0; k0 < C; k0 += 32) {
    __builtin_prefetch(A + (size_t)r0 * C + k0 + 64, 0, 1);
    v16h a0 = frag_a(A + (size_t)r0 * C + k0, C);
    v16h a1 = frag_a(A + (size_t)(r0 + 16) * C + k0, C);
    v16h b0 = frag_b(Wt + (size_t)c0 * C + k0, C);
    v16h b1 = frag_b(Wt + (size_t)(c0 + 16) * C + k0, C);
    acc[0][0] = wmma_f16(a0, b0, acc[0][0]);
    acc[0][1] = wmma_f16(a0, b1, acc[0][1]);
    acc[1][0] = wmma_f16(a1, b0, acc[1][0]);
    acc[1][1] = wmma_f16(a1, b1, acc[1][1]);
  }

  int col = lane & 15, hl = lane >> 4;
  if (mode == 2) {
    _Float16* V = v_cn + (size_t)b * C * N;
#pragma unroll
    for (int i = 0; i < 2; ++i)
#pragma unroll
      for (int j = 0; j < 2; ++j) {
        int cc = c0 + j * 16 + col;
        float bv = bias[cc];
#pragma unroll
        for (int r = 0; r < 8; ++r)
          V[(size_t)cc * N + r0 + i * 16 + r + 8 * hl] = (_Float16)(acc[i][j][r] + bv);
      }
  } else {
    _Float16* D = ((mode == 0) ? q_t : k_t) + (size_t)b * N * C;
    float sc = (mode == 0) ? QSCALE : 1.f;
#pragma unroll
    for (int i = 0; i < 2; ++i)
#pragma unroll
      for (int j = 0; j < 2; ++j) {
        int cc = c0 + j * 16 + col;
        float bv = bias[cc];
#pragma unroll
        for (int r = 0; r < 8; ++r)
          D[(size_t)(r0 + i * 16 + r + 8 * hl) * C + cc] = (_Float16)((acc[i][j][r] + bv) * sc);
      }
  }
}

// ---- Fused two-phase flash attention ---------------------------------------
// Block: 16 query rows, 8 waves. Q tile (16x512 f16 = 16KB contiguous) is
// DMA'd once into LDS by the Tensor Data Mover; both phases read A-fragments
// from LDS, removing the 8x redundant global Q traffic.
__global__ void __launch_bounds__(256)
attn_kernel(const _Float16* q_t, const _Float16* k_t, const _Float16* v_cn,
            _Float16* o_t) {
  constexpr int PLD = 136;  // 128 + 8-half pad, keeps 16B row alignment
  __shared__ __align__(16) _Float16 Qs[16 * C];   // 16 KB
  __shared__ _Float16 P[16 * PLD];
  __shared__ float wmax[8][16], wsum[8][16];
  __shared__ float rowM[16], rowInvL[16];

  int b  = blockIdx.x / (N / 16);
  int q0 = (blockIdx.x % (N / 16)) * 16;
  int wid = threadIdx.x >> 5, lane = threadIdx.x & 31;
  int col = lane & 15, hl = lane >> 4;
  const _Float16* Q = q_t + (size_t)b * N * C + (size_t)q0 * C;
  const _Float16* K = k_t + (size_t)b * N * C;
  const _Float16* V = v_cn + (size_t)b * C * N;

  // ---- stage Q tile into LDS via TDM (one contiguous 16KB transfer) ----
#if __has_builtin(__builtin_amdgcn_tensor_load_to_lds)
  if (wid == 0) {
    unsigned lds_off = (unsigned)(size_t)(void*)Qs;       // LDS byte offset
    unsigned long long ga = (unsigned long long)(const void*)Q;
    // D# group0: count=1 | lds_addr | global_addr[56:0] | type=2
    u32x4 g0 = { 1u,
                 lds_off,
                 (unsigned)(ga & 0xFFFFFFFFu),
                 (unsigned)((ga >> 32) & 0x01FFFFFFu) | (2u << 30) };
    // D# group1: data_size=3 (8B), tensor_dim0=2048, tile_dim0=2048,
    //            tensor_dim0_stride=2048 (1-D contiguous copy)
    i32x8 g1 = { (int)(3u << 16),
                 (int)(2048u << 16),   // tensor_dim0 -> bits[79:48]
                 0,
                 (int)(2048u << 16),   // tile_dim0   -> bits[127:112]
                 0,
                 2048,                 // tensor_dim0_stride -> bits[207:160]
                 0, 0 };
    i32x4 z4 = {0, 0, 0, 0};
#if defined(__clang_major__) && __clang_major__ >= 23
    i32x8 z8 = {0, 0, 0, 0, 0, 0, 0, 0};
    __builtin_amdgcn_tensor_load_to_lds(g0, g1, z4, z4, z8, 0);
#else
    __builtin_amdgcn_tensor_load_to_lds(g0, g1, z4, z4, 0);
#endif
    __builtin_amdgcn_s_wait_tensorcnt(0);
  }
  __syncthreads();
#else
  for (int i = threadIdx.x; i < 16 * C / 8; i += 256)
    ((v8h*)Qs)[i] = ((const v8h*)Q)[i];
  __syncthreads();
#endif

  float mx[8], sm[8];
#pragma unroll
  for (int r = 0; r < 8; ++r) { mx[r] = -INFINITY; sm[r] = 0.f; }

  // ---- Phase A: softmax statistics (online max / sumexp) ----
  for (int j = wid; j < N / 16; j += 8) {
    v8f s = vzero8();
#pragma unroll 4
    for (int k0 = 0; k0 < C; k0 += 32) {
      v16h af = frag_a(Qs + k0, C);                       // ds_load_b128
      v16h bf = frag_b(K + (size_t)j * 16 * C + k0, C);
      s = wmma_f16(af, bf, s);
    }
#pragma unroll
    for (int r = 0; r < 8; ++r) {
      float sv = s[r];
      float tm = sv;
      for (int o = 1; o < 16; o <<= 1) tm = fmaxf(tm, __shfl_xor(tm, o, 32));
      float nm = fmaxf(mx[r], tm);
      float e = __expf(sv - nm);
      for (int o = 1; o < 16; o <<= 1) e += __shfl_xor(e, o, 32);
      sm[r] = sm[r] * __expf(mx[r] - nm) + e;
      mx[r] = nm;
    }
  }
  if (col == 0) {
#pragma unroll
    for (int r = 0; r < 8; ++r) { wmax[wid][r + 8 * hl] = mx[r]; wsum[wid][r + 8 * hl] = sm[r]; }
  }
  __syncthreads();
  if (threadIdx.x < 16) {
    int t = threadIdx.x;
    float M = -INFINITY;
    for (int ww = 0; ww < 8; ++ww) M = fmaxf(M, wmax[ww][t]);
    float L = 0.f;
    for (int ww = 0; ww < 8; ++ww) L += wsum[ww][t] * __expf(wmax[ww][t] - M);
    rowM[t] = M; rowInvL[t] = 1.f / L;
  }
  __syncthreads();
  float Mr[8], IL[8];
#pragma unroll
  for (int r = 0; r < 8; ++r) { Mr[r] = rowM[r + 8 * hl]; IL[r] = rowInvL[r + 8 * hl]; }

  // ---- Phase B: O = P x V, wave owns 64 output channels ----
  int cw = wid * 64;
  v8f acc[4] = {vzero8(), vzero8(), vzero8(), vzero8()};
  for (int kb = 0; kb < N; kb += 128) {
    v8f s = vzero8();
    const _Float16* Kp = K + (size_t)(kb + wid * 16) * C;
#pragma unroll 4
    for (int k0 = 0; k0 < C; k0 += 32) {
      v16h af = frag_a(Qs + k0, C);
      v16h bf = frag_b(Kp + k0, C);
      s = wmma_f16(af, bf, s);
    }
#pragma unroll
    for (int r = 0; r < 8; ++r) {
      float p = __expf(s[r] - Mr[r]);
      P[(r + 8 * hl) * PLD + wid * 16 + col] = (_Float16)p;
    }
    __syncthreads();
#pragma unroll
    for (int km = 0; km < 4; ++km) {
      int akb = km * 32 + (hl << 3);
      v8h lo = *(const v8h*)&P[col * PLD + akb];
      v8h hi = *(const v8h*)&P[col * PLD + akb + 16];
      v16h af;
#pragma unroll
      for (int i = 0; i < 8; ++i) { af[i] = lo[i]; af[i + 8] = hi[i]; }
#pragma unroll
      for (int t = 0; t < 4; ++t) {
        v16h bf = frag_b(V + (size_t)(cw + t * 16) * N + kb + km * 32, N);
        acc[t] = wmma_f16(af, bf, acc[t]);
      }
    }
    __syncthreads();
  }

  _Float16* O = o_t + (size_t)b * N * C + (size_t)q0 * C;
#pragma unroll
  for (int t = 0; t < 4; ++t)
#pragma unroll
    for (int r = 0; r < 8; ++r)
      O[(size_t)(r + 8 * hl) * C + cw + t * 16 + col] = (_Float16)(acc[t][r] * IL[r]);
}

// ---- Projection GEMM + bias + residual, stores final f32 output ------------
__global__ void __launch_bounds__(256)
proj_gemm_kernel(const _Float16* o_t, const _Float16* wp, const float* pb,
                 const float* x, float* out) {
  int b = blockIdx.y;
  int bm = blockIdx.x / (C / 64), bn = blockIdx.x % (C / 64);
  int wid = threadIdx.x >> 5, lane = threadIdx.x & 31;
  int wm = wid & 3, wn = wid >> 2;
  int r0 = bm * 128 + wm * 32;
  int c0 = bn * 64  + wn * 32;
  const _Float16* A = o_t + (size_t)b * N * C;

  v8f acc[2][2] = {{vzero8(), vzero8()}, {vzero8(), vzero8()}};
#pragma unroll 4
  for (int k0 = 0; k0 < C; k0 += 32) {
    __builtin_prefetch(A + (size_t)r0 * C + k0 + 64, 0, 1);
    v16h a0 = frag_a(A + (size_t)r0 * C + k0, C);
    v16h a1 = frag_a(A + (size_t)(r0 + 16) * C + k0, C);
    v16h b0 = frag_b(wp + (size_t)c0 * C + k0, C);
    v16h b1 = frag_b(wp + (size_t)(c0 + 16) * C + k0, C);
    acc[0][0] = wmma_f16(a0, b0, acc[0][0]);
    acc[0][1] = wmma_f16(a0, b1, acc[0][1]);
    acc[1][0] = wmma_f16(a1, b0, acc[1][0]);
    acc[1][1] = wmma_f16(a1, b1, acc[1][1]);
  }

  int col = lane & 15, hl = lane >> 4;
  const float* xb = x + (size_t)b * C * N;
  float* ob = out + (size_t)b * C * N;
#pragma unroll
  for (int i = 0; i < 2; ++i)
#pragma unroll
    for (int j = 0; j < 2; ++j) {
      int cc = c0 + j * 16 + col;
      float bv = pb[cc];
#pragma unroll
      for (int r = 0; r < 8; ++r) {
        size_t idx = (size_t)cc * N + r0 + i * 16 + r + 8 * hl;
        ob[idx] = xb[idx] + acc[i][j][r] + bv;
      }
    }
}

// ---- host launch ------------------------------------------------------------
extern "C" void kernel_launch(void* const* d_in, const int* in_sizes, int n_in,
                              void* d_out, int out_size, void* d_ws, size_t ws_size,
                              hipStream_t stream) {
  (void)in_sizes; (void)n_in; (void)out_size; (void)ws_size;
  const float* x     = (const float*)d_in[0];
  const float* gamma = (const float*)d_in[1];
  const float* beta  = (const float*)d_in[2];
  const float* qw    = (const float*)d_in[3];
  const float* qb    = (const float*)d_in[4];
  const float* kw    = (const float*)d_in[5];
  const float* kbias = (const float*)d_in[6];
  const float* vw    = (const float*)d_in[7];
  const float* vb    = (const float*)d_in[8];
  const float* pw    = (const float*)d_in[9];
  const float* pb    = (const float*)d_in[10];
  float* out = (float*)d_out;

  char* ws = (char*)d_ws;
  size_t off = 0;
  auto wsalloc = [&](size_t bytes) -> void* {
    void* p = ws + off;
    off += (bytes + 255) & ~(size_t)255;
    return p;
  };
  _Float16* y_t  = (_Float16*)wsalloc((size_t)BB * N * C * sizeof(_Float16));
  _Float16* q_t  = (_Float16*)wsalloc((size_t)BB * N * C * sizeof(_Float16));
  _Float16* k_t  = (_Float16*)wsalloc((size_t)BB * N * C * sizeof(_Float16));
  _Float16* v_cn = (_Float16*)wsalloc((size_t)BB * C * N * sizeof(_Float16));
  _Float16* o_t  = (_Float16*)wsalloc((size_t)BB * N * C * sizeof(_Float16));
  _Float16* wq_h = (_Float16*)wsalloc((size_t)C * C * sizeof(_Float16));
  _Float16* wk_h = (_Float16*)wsalloc((size_t)C * C * sizeof(_Float16));
  _Float16* wv_h = (_Float16*)wsalloc((size_t)C * C * sizeof(_Float16));
  _Float16* wp_h = (_Float16*)wsalloc((size_t)C * C * sizeof(_Float16));

  prep_weights_kernel<<<dim3((C * C + 255) / 256), 256, 0, stream>>>(
      qw, kw, vw, pw, wq_h, wk_h, wv_h, wp_h);

  groupnorm_kernel<<<dim3(BB * G), 256, 0, stream>>>(x, gamma, beta, y_t);

  qkv_gemm_kernel<<<dim3((N / 128) * (C / 64), BB, 3), 256, 0, stream>>>(
      y_t, wq_h, wk_h, wv_h, qb, kbias, vb, q_t, k_t, v_cn);

  attn_kernel<<<dim3(BB * (N / 16)), 256, 0, stream>>>(q_t, k_t, v_cn, o_t);

  proj_gemm_kernel<<<dim3((N / 128) * (C / 64), BB), 256, 0, stream>>>(
      o_t, wp_h, pb, x, out);
}
